// EncodeProcessDecode_8924942041889
// MI455X (gfx1250) — compile-verified, matrix operand order checked by reading, not run on hardware
//
#include <hip/hip_runtime.h>
#include <hip/hip_bf16.h>
#include <math.h>

typedef _Float16 half_t;
typedef __attribute__((ext_vector_type(16))) _Float16 v16h;
typedef __attribute__((ext_vector_type(8)))  _Float16 v8h;
typedef __attribute__((ext_vector_type(8)))  float    v8f;

#define N_NODES 20000
#define N_EDGES 320000

// ---------------------------------------------------------------------------
// A-operand descriptor: plain f16 rows, f32 rows (padded), or fused gathers.
// ---------------------------------------------------------------------------
struct GemmA {
    const half_t* a16; int lda;     // MODE 0
    const float*  a32; int lda32;   // MODE 4
    int Keff;                       // valid K (cols >= Keff are zero-padding)
    const half_t* nodeL;            // node latents [N,128] f16
    const half_t* edgeL;            // edge latents / edge_k [E,128] f16
    const half_t* aggL;             // aggregated messages [N,128] f16
    const float*  u;                // globals [8,16] f32
    const int* row; const int* col; const int* batch;
};

struct GemmP {
    GemmA A;
    const half_t* Bfrag;   // [KC][8 tiles][32 lanes][16 halves] f16 fragments
    const float* bias;     // [128]
    int M;
    const float* g; const float* beta;   // LayerNorm affine (EPI==1)
    half_t* out;           // [M,128] f16
    const half_t* prev;    // residual input  (RES)
    half_t* out2;          // residual output (RES)
};

__device__ inline v8h cvt8(const float* p) {
    v8h o;
    #pragma unroll
    for (int i = 0; i < 8; ++i) o[i] = (half_t)p[i];
    return o;
}

// MODE: 0=f16 plain, 1=edge concat, 2=msg concat, 3=node concat, 4=f32 plain
template<int MODE>
__device__ inline v8h load8(const GemmA& A, int r, int c) {
    v8h z = {};
    if (c >= A.Keff) return z;
    if (MODE == 0)
        return *(const v8h*)(A.a16 + (size_t)r * A.lda + c);
    if (MODE == 4)
        return cvt8(A.a32 + (size_t)r * A.lda32 + c);
    if (MODE == 1) {  // [node[row] | node[col] | edge | u[batch[row]] | pad]
        if (c < 128)      return *(const v8h*)(A.nodeL + (size_t)A.row[r] * 128 + c);
        else if (c < 256) return *(const v8h*)(A.nodeL + (size_t)A.col[r] * 128 + (c - 128));
        else if (c < 384) return *(const v8h*)(A.edgeL + (size_t)r * 128 + (c - 256));
        else              return cvt8(A.u + (size_t)A.batch[A.row[r]] * 16 + (c - 384));
    }
    if (MODE == 2) {  // [node[row] | edge_k]
        if (c < 128) return *(const v8h*)(A.nodeL + (size_t)A.row[r] * 128 + c);
        else         return *(const v8h*)(A.edgeL + (size_t)r * 128 + (c - 128));
    }
    if (MODE == 3) {  // [node | agg | u[batch] | pad]
        if (c < 128)      return *(const v8h*)(A.nodeL + (size_t)r * 128 + c);
        else if (c < 256) return *(const v8h*)(A.aggL + (size_t)r * 128 + (c - 128));
        else              return cvt8(A.u + (size_t)A.batch[r] * 16 + (c - 256));
    }
    return z;
}

// ---------------------------------------------------------------------------
// GEMM: out[M,128] = epilogue(A[M,K] @ W[K,128] + bias)
// Block = 256 threads = 8 waves; wave = 16 rows x 128 cols (8 WMMA tiles).
// EPI: 0 = bias+ELU, 1 = bias+LayerNorm. RES: also write prev+y to out2.
// ---------------------------------------------------------------------------
template<int MODE, int EPI, bool RES, int KC>
__global__ __launch_bounds__(256) void gemm128_wmma(GemmP P) {
    __shared__ float sacc[8][16][128];   // 64 KB
    __shared__ float smu[128], sinv[128];

    const int lane = threadIdx.x & 31;
    const int w    = threadIdx.x >> 5;
    const bool hi  = lane >= 16;

    int r = blockIdx.x * 128 + w * 16 + (lane & 15);
    if (r >= P.M) r = P.M - 1;           // clamp; stores masked later

    v8f acc[8] = {};

    #pragma unroll
    for (int kc = 0; kc < KC; ++kc) {
        const int kb = kc * 32;
        // A fragment per ISA 16-bit 16x32 layout: lanes 0-15 -> K {0..7,16..23},
        // lanes 16-31 -> K {8..15,24..31}, row = lane&15.
        v8h g0 = load8<MODE>(P.A, r, kb + (hi ? 8 : 0));
        v8h g1 = load8<MODE>(P.A, r, kb + (hi ? 24 : 16));
        v16h a;
        #pragma unroll
        for (int i = 0; i < 8; ++i) { a[i] = g0[i]; a[8 + i] = g1[i]; }

        if (kc + 1 < KC)
            __builtin_prefetch(P.Bfrag + (size_t)(kc + 1) * 4096, 0, 1);

        const v16h* bf = (const v16h*)P.Bfrag + (size_t)kc * 256;
        #pragma unroll
        for (int t = 0; t < 8; ++t) {
            v16h b = bf[t * 32 + lane];
            acc[t] = __builtin_amdgcn_wmma_f32_16x16x32_f16(
                false, a, false, b, (short)0, acc[t], false, false);
        }
    }

    // Spill accumulators (C/D layout: vgpr q, lanes 0-15 -> row q, 16-31 -> row q+8)
    {
        const int cc = (lane & 15);
        const int rb = hi ? 8 : 0;
        #pragma unroll
        for (int t = 0; t < 8; ++t) {
            #pragma unroll
            for (int q = 0; q < 8; ++q)
                sacc[w][rb + q][t * 16 + cc] = acc[t][q];
        }
    }
    __syncthreads();

    const int rl = threadIdx.x & 127;
    const int gr = blockIdx.x * 128 + rl;
    const float* srow = sacc[rl >> 4][rl & 15];

    if (EPI == 1) {   // per-row LayerNorm stats (one thread per row)
        if (threadIdx.x < 128 && gr < P.M) {
            float s = 0.f, s2 = 0.f;
            for (int c = 0; c < 128; ++c) {
                float v = srow[c] + P.bias[c];
                s += v; s2 += v * v;
            }
            const float mu  = s * (1.f / 128.f);
            const float var = s2 * (1.f / 128.f) - mu * mu;
            smu[rl]  = mu;
            sinv[rl] = rsqrtf(var + 1e-5f);
        }
        __syncthreads();
    }

    if (gr < P.M) {  // all 256 threads: half a row each, vectorized stores
        const int c0 = (threadIdx.x >> 7) * 64;
        if (EPI == 0) {
            #pragma unroll
            for (int cb = 0; cb < 64; cb += 8) {
                v8h o;
                #pragma unroll
                for (int i = 0; i < 8; ++i) {
                    const int c = c0 + cb + i;
                    float v = srow[c] + P.bias[c];
                    v = v > 0.f ? v : (__expf(v) - 1.f);
                    o[i] = (half_t)v;
                }
                *(v8h*)(P.out + (size_t)gr * 128 + c0 + cb) = o;
            }
        } else {
            const float mu = smu[rl], inv = sinv[rl];
            #pragma unroll
            for (int cb = 0; cb < 64; cb += 8) {
                v8h o, o2;
                v8h pv = {};
                if (RES)
                    pv = *(const v8h*)(P.prev + (size_t)gr * 128 + c0 + cb);
                #pragma unroll
                for (int i = 0; i < 8; ++i) {
                    const int c = c0 + cb + i;
                    float v = (srow[c] + P.bias[c] - mu) * inv * P.g[c] + P.beta[c];
                    o[i] = (half_t)v;
                    if (RES) o2[i] = (half_t)((float)pv[i] + v);
                }
                *(v8h*)(P.out + (size_t)gr * 128 + c0 + cb) = o;
                if (RES)
                    *(v8h*)(P.out2 + (size_t)gr * 128 + c0 + cb) = o2;
            }
        }
    }
}

// ---------------------------------------------------------------------------
// Weight prep: f32 [Keff,128] row-major -> f16 WMMA B fragments
// (k-grouping mirrors the A-fragment layout; zero-pads k >= Keff).
// ---------------------------------------------------------------------------
__global__ void prep_frag(const float* __restrict__ W, half_t* __restrict__ dst,
                          int Keff, int KC) {
    const int idx = blockIdx.x * blockDim.x + threadIdx.x;
    const int total = KC * 4096;
    if (idx >= total) return;
    const int h    = idx & 15;
    const int lane = (idx >> 4) & 31;
    const int t    = (idx >> 9) & 7;
    const int kc   = idx >> 12;
    const bool hiL = lane >= 16;
    const int k = kc * 32 + (hiL ? (h < 8 ? h + 8 : h + 16)
                                 : (h < 8 ? h     : h + 8));
    const int n = t * 16 + (lane & 15);
    half_t v = (half_t)0.f;
    if (k < Keff) v = (half_t)W[(size_t)k * 128 + n];
    dst[idx] = v;
}

// ---------------------------------------------------------------------------
// Scatter-mean helpers
// ---------------------------------------------------------------------------
__global__ void zero_f32(float* p, size_t n) {
    size_t i = (size_t)blockIdx.x * blockDim.x + threadIdx.x;
    if (i < n) p[i] = 0.f;
}

__global__ void scatter_add(const half_t* __restrict__ msg, const int* __restrict__ col,
                            float* agg, float* cnt, int E) {
    const size_t idx = (size_t)blockIdx.x * blockDim.x + threadIdx.x;
    const int e = (int)(idx >> 7);
    const int c = (int)(idx & 127);
    if (e >= E) return;
    const int n = col[e];
    atomicAdd(&agg[(size_t)n * 128 + c], (float)msg[(size_t)e * 128 + c]);
    if (c == 0) atomicAdd(&cnt[n], 1.0f);
}

__global__ void normalize_agg(const float* __restrict__ agg, const float* __restrict__ cnt,
                              half_t* out, int N) {
    const size_t idx = (size_t)blockIdx.x * blockDim.x + threadIdx.x;
    const int n = (int)(idx >> 7);
    if (n >= N) return;
    float d = cnt[n];
    d = d < 1.f ? 1.f : d;
    out[idx] = (half_t)(agg[idx] / d);
}

// ---------------------------------------------------------------------------
// Final decoder layer: 128 -> 3, f32 out
// ---------------------------------------------------------------------------
__global__ void dec_final(const half_t* __restrict__ h, const float* __restrict__ W,
                          const float* __restrict__ b, float* __restrict__ out, int N) {
    const int n = blockIdx.x * blockDim.x + threadIdx.x;
    if (n >= N) return;
    float a0 = b[0], a1 = b[1], a2 = b[2];
    for (int k = 0; k < 128; ++k) {
        const float hv = (float)h[(size_t)n * 128 + k];
        a0 += hv * W[k * 3 + 0];
        a1 += hv * W[k * 3 + 1];
        a2 += hv * W[k * 3 + 2];
    }
    out[n * 3 + 0] = a0; out[n * 3 + 1] = a1; out[n * 3 + 2] = a2;
}

// ---------------------------------------------------------------------------
// Host helpers
// ---------------------------------------------------------------------------
static GemmP mkP(const GemmA& A, const half_t* Bf, const float* bias, int M,
                 const float* g, const float* beta, half_t* out,
                 const half_t* prev, half_t* out2) {
    GemmP P;
    P.A = A; P.Bfrag = Bf; P.bias = bias; P.M = M;
    P.g = g; P.beta = beta; P.out = out; P.prev = prev; P.out2 = out2;
    return P;
}

template<int MODE, int EPI, bool RES, int KC>
static void run_gemm(const GemmP& P, hipStream_t stream) {
    gemm128_wmma<MODE, EPI, RES, KC><<<(P.M + 127) / 128, 256, 0, stream>>>(P);
}

// ---------------------------------------------------------------------------
// Host orchestration
// ---------------------------------------------------------------------------
extern "C" void kernel_launch(void* const* d_in, const int* in_sizes, int n_in,
                              void* d_out, int out_size, void* d_ws, size_t ws_size,
                              hipStream_t stream) {
    (void)in_sizes; (void)n_in; (void)out_size; (void)ws_size;

    const int N = N_NODES, E = N_EDGES;
    const float* x    = (const float*)d_in[0];
    const float* eatt = (const float*)d_in[1];
    const int*   ei   = (const int*)d_in[2];
    const float* u    = (const float*)d_in[3];
    const int*   bat  = (const int*)d_in[4];
    const int* rowI = ei;
    const int* colI = ei + E;

    // param leaf indices (sorted-key flatten of params):
    //  dec:      5..10  (b0,b1,b2,W0,W1,W2)
    //  edge_enc: 11..18 (b0,b1,b2,g,beta,W0,W1,W2)
    //  node_enc: 19..26
    //  steps[s]: base=27+24*s : edge_mlp[base..base+7], node_mlp1[+8], node_mlp2[+16]
    auto F32 = [&](int i) { return (const float*)d_in[i]; };

    size_t off = 0;
    auto alloc = [&](size_t bytes) -> void* {
        off = (off + 255) & ~(size_t)255;
        void* p = (char*)d_ws + off;
        off += bytes;
        return p;
    };

    auto prep = [&](int widx, int Keff) -> half_t* {
        const int KC = (Keff + 31) / 32;
        const size_t total = (size_t)KC * 4096;
        half_t* dst = (half_t*)alloc(total * sizeof(half_t));
        prep_frag<<<(int)((total + 255) / 256), 256, 0, stream>>>(F32(widx), dst, Keff, KC);
        return dst;
    };

    // --- weight fragments (~1.1 MB total) ---
    half_t* Fne[3] = { prep(24, 16),  prep(25, 128), prep(26, 128) };
    half_t* Fee[3] = { prep(16, 8),   prep(17, 128), prep(18, 128) };
    half_t* Fd [2] = { prep(8, 128),  prep(9, 128) };
    half_t *Fem[2][3], *Fm1[2][3], *Fm2[2][3];
    for (int s = 0; s < 2; ++s) {
        const int b = 27 + 24 * s;
        Fem[s][0] = prep(b + 5, 400);  Fem[s][1] = prep(b + 6, 128);  Fem[s][2] = prep(b + 7, 128);
        Fm1[s][0] = prep(b + 13, 256); Fm1[s][1] = prep(b + 14, 128); Fm1[s][2] = prep(b + 15, 128);
        Fm2[s][0] = prep(b + 21, 272); Fm2[s][1] = prep(b + 22, 128); Fm2[s][2] = prep(b + 23, 128);
    }

    // --- activation buffers (~450 MB) ---
    half_t* nodeA = (half_t*)alloc((size_t)N * 128 * 2);
    half_t* nodeB = (half_t*)alloc((size_t)N * 128 * 2);
    half_t* edgeA = (half_t*)alloc((size_t)E * 128 * 2);
    half_t* edgeB = (half_t*)alloc((size_t)E * 128 * 2);
    half_t* edgeK = (half_t*)alloc((size_t)E * 128 * 2);
    half_t* h1    = (half_t*)alloc((size_t)E * 128 * 2);
    half_t* h2    = (half_t*)alloc((size_t)E * 128 * 2);
    float*  agg   = (float*)alloc((size_t)N * 128 * 4);
    float*  cnt   = (float*)alloc((size_t)N * 4);
    half_t* aggH  = (half_t*)alloc((size_t)N * 128 * 2);

    auto plainA = [&](const half_t* a) {
        GemmA g = {}; g.a16 = a; g.lda = 128; g.Keff = 128; return g;
    };

    // ===== Encoders =====
    { GemmA A = {}; A.a32 = x; A.lda32 = 16; A.Keff = 16;
      run_gemm<4,0,false,1>(mkP(A, Fne[0], F32(19), N, nullptr, nullptr, h1, nullptr, nullptr), stream); }
    run_gemm<0,0,false,4>(mkP(plainA(h1), Fne[1], F32(20), N, nullptr, nullptr, h2, nullptr, nullptr), stream);
    run_gemm<0,1,false,4>(mkP(plainA(h2), Fne[2], F32(21), N, F32(22), F32(23), nodeA, nullptr, nullptr), stream);

    { GemmA A = {}; A.a32 = eatt; A.lda32 = 8; A.Keff = 8;
      run_gemm<4,0,false,1>(mkP(A, Fee[0], F32(11), E, nullptr, nullptr, h1, nullptr, nullptr), stream); }
    run_gemm<0,0,false,4>(mkP(plainA(h1), Fee[1], F32(12), E, nullptr, nullptr, h2, nullptr, nullptr), stream);
    run_gemm<0,1,false,4>(mkP(plainA(h2), Fee[2], F32(13), E, F32(14), F32(15), edgeA, nullptr, nullptr), stream);

    // ===== Message-passing steps =====
    for (int s = 0; s < 2; ++s) {
        const int b = 27 + 24 * s;
        const half_t* nodeCur = (s == 0) ? nodeA : nodeB;
        half_t*       nodeNxt = (s == 0) ? nodeB : nodeA;
        const half_t* edgeCur = (s == 0) ? edgeA : edgeB;
        half_t*       edgeNxt = (s == 0) ? edgeB : edgeA;

        // edge_mlp: [node[row]|node[col]|edge|u[batch[row]]] (400 -> pad 416)
        { GemmA A = {}; A.Keff = 400; A.nodeL = nodeCur; A.edgeL = edgeCur;
          A.u = u; A.row = rowI; A.col = colI; A.batch = bat;
          run_gemm<1,0,false,13>(mkP(A, Fem[s][0], F32(b+0), E, nullptr, nullptr, h1, nullptr, nullptr), stream); }
        run_gemm<0,0,false,4>(mkP(plainA(h1), Fem[s][1], F32(b+1), E, nullptr, nullptr, h2, nullptr, nullptr), stream);
        run_gemm<0,1,true,4>(mkP(plainA(h2), Fem[s][2], F32(b+2), E, F32(b+3), F32(b+4), edgeK, edgeCur, edgeNxt), stream);

        // node_mlp1: [node[row]|edge_k] (256)
        { GemmA A = {}; A.Keff = 256; A.nodeL = nodeCur; A.edgeL = edgeK; A.row = rowI;
          run_gemm<2,0,false,8>(mkP(A, Fm1[s][0], F32(b+8), E, nullptr, nullptr, h1, nullptr, nullptr), stream); }
        run_gemm<0,0,false,4>(mkP(plainA(h1), Fm1[s][1], F32(b+9), E, nullptr, nullptr, h2, nullptr, nullptr), stream);
        run_gemm<0,1,false,4>(mkP(plainA(h2), Fm1[s][2], F32(b+10), E, F32(b+11), F32(b+12), h1, nullptr, nullptr), stream); // msg -> h1

        // scatter-mean msg over col
        zero_f32<<<(int)(((size_t)N * 128 + 255) / 256), 256, 0, stream>>>(agg, (size_t)N * 128);
        zero_f32<<<(N + 255) / 256, 256, 0, stream>>>(cnt, (size_t)N);
        scatter_add<<<(int)(((size_t)E * 128 + 255) / 256), 256, 0, stream>>>(h1, colI, agg, cnt, E);
        normalize_agg<<<(int)(((size_t)N * 128 + 255) / 256), 256, 0, stream>>>(agg, cnt, aggH, N);

        // node_mlp2: [node|agg|u[batch]] (272 -> pad 288), residual into nodeNxt
        { GemmA A = {}; A.Keff = 272; A.nodeL = nodeCur; A.aggL = aggH;
          A.u = u; A.batch = bat;
          run_gemm<3,0,false,9>(mkP(A, Fm2[s][0], F32(b+16), N, nullptr, nullptr, h2, nullptr, nullptr), stream); }
        run_gemm<0,0,false,4>(mkP(plainA(h2), Fm2[s][1], F32(b+17), N, nullptr, nullptr, h1, nullptr, nullptr), stream);
        run_gemm<0,1,true,4>(mkP(plainA(h1), Fm2[s][2], F32(b+18), N, F32(b+19), F32(b+20), h2, nodeCur, nodeNxt), stream);
    }

    // ===== Decoder ===== (final node latents are in nodeA after 2 steps)
    run_gemm<0,0,false,4>(mkP(plainA(nodeA), Fd[0], F32(5), N, nullptr, nullptr, h1, nullptr, nullptr), stream);
    run_gemm<0,0,false,4>(mkP(plainA(h1),    Fd[1], F32(6), N, nullptr, nullptr, h2, nullptr, nullptr), stream);
    dec_final<<<(N + 255) / 256, 256, 0, stream>>>(h2, F32(10), F32(7), (float*)d_out, N);
}
